// LSTMDPGeneral_3100966388316
// MI455X (gfx1250) — matrix operand order repaired and checked by reading, not compiled, not run on hardware
//
#include <hip/hip_runtime.h>
#include <hip/hip_bf16.h>

// ---------------------------------------------------------------------------
// IBP-LSTM on gfx1250: bf16 WMMA GEMMs, weights staged in LDS, cell state in
// VGPRs, persistent kernel with one grid barrier per timestep.
// ---------------------------------------------------------------------------

typedef __bf16 bf16;
typedef __attribute__((ext_vector_type(8)))  bf16  v8bf;
typedef __attribute__((ext_vector_type(16))) bf16  v16bf;
typedef __attribute__((ext_vector_type(8)))  float v8f;

#define BB   64
#define TT   512
#define DD   128
#define HH   256
#define G4H  1024
#define NBLK 16

#define WIHV (G4H * DD)      // 131072 elements per W_ih variant
#define WHHV (G4H * HH)      // 262144 elements per W_hh variant
#define XV   (BB * TT * DD)  // 4194304 elements per x variant
#define HV   (BB * HH)       // 16384 elements per h variant
#define HBUF (3 * HV)        // one h triple buffer
#define BTH  (BB * TT * HH)  // 8388608 elements per output slab

// workspace byte offsets (all 256-aligned)
#define OFF_BAR 0
#define OFF_WIH 256
#define OFF_WHH (OFF_WIH + 3 * WIHV * 2)            // +786432
#define OFF_X   (OFF_WHH + 3 * WHHV * 2)            // +1572864
#define OFF_H   (OFF_X   + 3 * (long)XV * 2)        // +25165824
#define SMEM_BYTES ((3 * 64 * HH + 3 * 64 * DD) * 2)  // 147456

__device__ inline float sigm(float x) { return 1.0f / (1.0f + __expf(-x)); }
__device__ inline float tanha(float x) { return 1.0f - 2.0f / (__expf(2.0f * x) + 1.0f); }

__device__ inline v16bf cat16(v8bf lo, v8bf hi) {
  return __builtin_shufflevector(lo, hi, 0, 1, 2, 3, 4, 5, 6, 7, 8, 9, 10, 11, 12, 13, 14, 15);
}
// A-fragment (16x32 bf16): lane holds row m=(lane&15); 8 bf16 at kbase and 8 at kbase+16
__device__ inline v16bf ld_afrag(const bf16* row, int kb) {
  v8bf lo = *(const v8bf*)(row + kb);
  v8bf hi = *(const v8bf*)(row + kb + 16);
  return cat16(lo, hi);
}
// B-fragment (32x16 bf16): lane holds col n=(lane&15); 8 bf16 at kbase and 8 at kbase+16
__device__ inline v16bf ld_bfrag(const bf16* p) {
  v8bf lo = *(const v8bf*)p;
  v8bf hi = *(const v8bf*)(p + 16);
  return cat16(lo, hi);
}

__device__ inline v8f wmma_bf16(v16bf a, v16bf b, v8f c) {
  return __builtin_amdgcn_wmma_f32_16x16x32_bf16(false, a, false, b, (short)0, c, false, false);
}

__device__ inline void grid_sync(unsigned* bar) {
  __threadfence();
  __syncthreads();
  if (threadIdx.x == 0) {
    unsigned ticket = __hip_atomic_fetch_add(&bar[0], 1u, __ATOMIC_ACQ_REL, __HIP_MEMORY_SCOPE_AGENT);
    unsigned target = ticket / NBLK + 1u;
    if (ticket % NBLK == NBLK - 1u)
      __hip_atomic_fetch_add(&bar[1], 1u, __ATOMIC_ACQ_REL, __HIP_MEMORY_SCOPE_AGENT);
    while (__hip_atomic_load(&bar[1], __ATOMIC_ACQUIRE, __HIP_MEMORY_SCOPE_AGENT) < target) {
      __builtin_amdgcn_s_sleep(2);
    }
  }
  __syncthreads();
}

// ---------------------------------------------------------------------------
// Prep: bf16 conversion of x triple, W_ih/W_hh split into val/pos/neg bf16,
// zero the double-buffered bf16 h state, reset the grid barrier.
// ---------------------------------------------------------------------------
__global__ void prep_kernel(const float* __restrict__ xv, const float* __restrict__ xl,
                            const float* __restrict__ xu, const float* __restrict__ wih,
                            const float* __restrict__ whh, bf16* __restrict__ wih_bf,
                            bf16* __restrict__ whh_bf, bf16* __restrict__ x_bf,
                            bf16* __restrict__ h_bf, unsigned* __restrict__ bar) {
  const long tid = (long)blockIdx.x * blockDim.x + threadIdx.x;
  const long stride = (long)gridDim.x * blockDim.x;
  for (long i = tid; i < WIHV; i += stride) {
    float w = wih[i];
    wih_bf[i] = (bf16)w;
    wih_bf[WIHV + i] = (bf16)fmaxf(w, 0.0f);
    wih_bf[2 * WIHV + i] = (bf16)fminf(w, 0.0f);
  }
  for (long i = tid; i < WHHV; i += stride) {
    float w = whh[i];
    whh_bf[i] = (bf16)w;
    whh_bf[WHHV + i] = (bf16)fmaxf(w, 0.0f);
    whh_bf[2 * WHHV + i] = (bf16)fminf(w, 0.0f);
  }
  for (long i = tid; i < XV; i += stride) {
    x_bf[i] = (bf16)xv[i];
    x_bf[XV + i] = (bf16)xl[i];
    x_bf[2L * XV + i] = (bf16)xu[i];
  }
  for (long i = tid; i < 2 * HBUF; i += stride) h_bf[i] = (bf16)0.0f;
  if (tid == 0) { bar[0] = 0u; bar[1] = 0u; }
}

// ---------------------------------------------------------------------------
// Persistent recurrent kernel. Grid = 16 blocks (one per j-tile of 16 hidden
// units) x 128 threads (4 waves = 4 M-tiles of 16 batch rows). Each wave keeps
// its interval cell state in registers and issues 240 bf16 WMMAs per step.
// ---------------------------------------------------------------------------
__global__ __launch_bounds__(128) void lstm_ibp_kernel(
    const float* __restrict__ bias, const bf16* __restrict__ wih_bf,
    const bf16* __restrict__ whh_bf, const bf16* __restrict__ x_bf,
    bf16* __restrict__ h_bf, float* __restrict__ out, unsigned* __restrict__ bar) {
  extern __shared__ char smem[];
  bf16* s_whh = (bf16*)smem;             // [3 variants][4 gates * 16 j][256]
  bf16* s_wih = s_whh + 3 * 64 * HH;     // [3 variants][4 gates * 16 j][128]

  const int tid = threadIdx.x;
  const int wave = tid >> 5;
  const int lane = tid & 31;
  const int jt = blockIdx.x;             // 0..15
  const int mt = wave;                   // 0..3
  const int ncol = lane & 15;
  const int ahalf = (lane >> 4) * 8;
  const int bhalf = (lane >> 4) * 8;

  // --- stage this block's weight rows (gate*256 + jt*16 + jloc) into LDS ---
  for (int q = tid; q < 3 * 64 * HH / 8; q += 128) {
    long e = (long)q * 8;
    int v = (int)(e / (64 * HH));
    int rem = (int)(e % (64 * HH));
    int grow = rem / HH;                  // 0..63 = gate*16 + jloc
    int k = rem % HH;
    int wr = (grow >> 4) * HH + jt * 16 + (grow & 15);
    *(uint4*)(s_whh + e) = *(const uint4*)(whh_bf + (long)v * WHHV + (long)wr * HH + k);
  }
  for (int q = tid; q < 3 * 64 * DD / 8; q += 128) {
    long e = (long)q * 8;
    int v = (int)(e / (64 * DD));
    int rem = (int)(e % (64 * DD));
    int grow = rem / DD;
    int k = rem % DD;
    int wr = (grow >> 4) * HH + jt * 16 + (grow & 15);
    *(uint4*)(s_wih + e) = *(const uint4*)(wih_bf + (long)v * WIHV + (long)wr * DD + k);
  }
  __syncthreads();

  float bb[4];
#pragma unroll
  for (int g = 0; g < 4; ++g) bb[g] = bias[g * HH + jt * 16 + ncol];

  const int mrow = mt * 16 + ncol;       // A-matrix row this lane feeds
  v8f cv, cl, cu;
#pragma unroll
  for (int e = 0; e < 8; ++e) { cv[e] = 0.0f; cl[e] = 0.0f; cu[e] = 0.0f; }

  for (int t = 0; t < TT; ++t) {
    const bf16* hsrc = h_bf + (t & 1) * HBUF;
    bf16* hdst = h_bf + ((t + 1) & 1) * HBUF;

    v8f acc[4][3];
#pragma unroll
    for (int g = 0; g < 4; ++g) {
      v8f s;
#pragma unroll
      for (int e = 0; e < 8; ++e) s[e] = bb[g];
      acc[g][0] = s; acc[g][1] = s; acc[g][2] = s;
    }

    // ---- h-path GEMMs: pre += h_triple @ {Wv,Wp,Wn}_hh^T ----
    const bf16* hr0 = hsrc + (long)mrow * HH;
    const bf16* hr1 = hr0 + HV;
    const bf16* hr2 = hr1 + HV;
    for (int kc = 0; kc < HH / 32; ++kc) {
      const int ka = kc * 32 + ahalf;
      v16bf av = ld_afrag(hr0, ka);
      v16bf al = ld_afrag(hr1, ka);
      v16bf au = ld_afrag(hr2, ka);
      const int kb = kc * 32 + bhalf;
#pragma unroll
      for (int g = 0; g < 4; ++g) {
        const bf16* wp = s_whh + ((g * 16 + ncol) * HH + kb);
        v16bf bv = ld_bfrag(wp);
        v16bf bp = ld_bfrag(wp + 64 * HH);
        v16bf bn = ld_bfrag(wp + 2 * 64 * HH);
        acc[g][0] = wmma_bf16(av, bv, acc[g][0]);
        acc[g][1] = wmma_bf16(al, bp, acc[g][1]);
        acc[g][1] = wmma_bf16(au, bn, acc[g][1]);
        acc[g][2] = wmma_bf16(au, bp, acc[g][2]);
        acc[g][2] = wmma_bf16(al, bn, acc[g][2]);
      }
    }

    // ---- x-path GEMMs: pre += x_triple(t) @ {Wv,Wp,Wn}_ih^T ----
    const bf16* xr0 = x_bf + (long)mrow * (TT * DD) + (long)t * DD;
    const bf16* xr1 = xr0 + XV;
    const bf16* xr2 = xr1 + XV;
    for (int kc = 0; kc < DD / 32; ++kc) {
      const int ka = kc * 32 + ahalf;
      v16bf av = ld_afrag(xr0, ka);
      v16bf al = ld_afrag(xr1, ka);
      v16bf au = ld_afrag(xr2, ka);
      const int kb = kc * 32 + bhalf;
#pragma unroll
      for (int g = 0; g < 4; ++g) {
        const bf16* wp = s_wih + ((g * 16 + ncol) * DD + kb);
        v16bf bv = ld_bfrag(wp);
        v16bf bp = ld_bfrag(wp + 64 * DD);
        v16bf bn = ld_bfrag(wp + 2 * 64 * DD);
        acc[g][0] = wmma_bf16(av, bv, acc[g][0]);
        acc[g][1] = wmma_bf16(al, bp, acc[g][1]);
        acc[g][1] = wmma_bf16(au, bn, acc[g][1]);
        acc[g][2] = wmma_bf16(au, bp, acc[g][2]);
        acc[g][2] = wmma_bf16(al, bn, acc[g][2]);
      }
    }

    // ---- gate activations + interval cell/hidden update (wave-local) ----
#pragma unroll
    for (int e = 0; e < 8; ++e) {
      float i_v = sigm(acc[0][0][e]), i_l = sigm(acc[0][1][e]), i_u = sigm(acc[0][2][e]);
      float f_v = sigm(acc[1][0][e]), f_l = sigm(acc[1][1][e]), f_u = sigm(acc[1][2][e]);
      float g_v = tanha(acc[2][0][e]), g_l = tanha(acc[2][1][e]), g_u = tanha(acc[2][2][e]);
      float o_v = sigm(acc[3][0][e]), o_l = sigm(acc[3][1][e]), o_u = sigm(acc[3][2][e]);

      float ccl = cl[e], ccu = cu[e];
      float p1 = f_l * ccl, p2 = f_l * ccu, p3 = f_u * ccl, p4 = f_u * ccu;
      float q1 = i_l * g_l, q2 = i_l * g_u, q3 = i_u * g_l, q4 = i_u * g_u;
      float ncv = f_v * cv[e] + i_v * g_v;
      float ncl = fminf(fminf(p1, p2), fminf(p3, p4)) + fminf(fminf(q1, q2), fminf(q3, q4));
      float ncu = fmaxf(fmaxf(p1, p2), fmaxf(p3, p4)) + fmaxf(fmaxf(q1, q2), fmaxf(q3, q4));
      cv[e] = ncv; cl[e] = ncl; cu[e] = ncu;

      float tv = tanha(ncv), tl = tanha(ncl), tu = tanha(ncu);
      float r1 = o_l * tl, r2 = o_l * tu, r3 = o_u * tl, r4 = o_u * tu;
      float h_v = o_v * tv;
      float h_l = fminf(fminf(r1, r2), fminf(r3, r4));
      float h_u = fmaxf(fmaxf(r1, r2), fmaxf(r3, r4));

      const int m = mt * 16 + ((lane < 16) ? e : e + 8);
      const int j = jt * 16 + ncol;
      const long ob = (long)m * (TT * HH) + (long)t * HH + j;
      out[ob] = h_v;
      out[(long)BTH + ob] = h_l;
      out[2L * BTH + ob] = h_u;
      const long hb = (long)m * HH + j;
      hdst[hb] = (bf16)h_v;
      hdst[HV + hb] = (bf16)h_l;
      hdst[2 * HV + hb] = (bf16)h_u;
    }

    grid_sync(bar);  // h(t) visible device-wide before step t+1
  }
}

extern "C" void kernel_launch(void* const* d_in, const int* in_sizes, int n_in,
                              void* d_out, int out_size, void* d_ws, size_t ws_size,
                              hipStream_t stream) {
  (void)in_sizes; (void)n_in; (void)out_size; (void)ws_size;
  const float* xv = (const float*)d_in[0];
  const float* xl = (const float*)d_in[1];
  const float* xu = (const float*)d_in[2];
  const float* wih = (const float*)d_in[3];
  const float* whh = (const float*)d_in[4];
  const float* bias = (const float*)d_in[5];
  float* out = (float*)d_out;

  char* ws = (char*)d_ws;
  unsigned* bar = (unsigned*)(ws + OFF_BAR);
  bf16* wih_bf = (bf16*)(ws + OFF_WIH);
  bf16* whh_bf = (bf16*)(ws + OFF_WHH);
  bf16* x_bf = (bf16*)(ws + OFF_X);
  bf16* h_bf = (bf16*)(ws + OFF_H);

  prep_kernel<<<512, 256, 0, stream>>>(xv, xl, xu, wih, whh, wih_bf, whh_bf, x_bf, h_bf, bar);

  hipFuncSetAttribute(reinterpret_cast<const void*>(lstm_ibp_kernel),
                      hipFuncAttributeMaxDynamicSharedMemorySize, SMEM_BYTES);
  lstm_ibp_kernel<<<NBLK, 128, SMEM_BYTES, stream>>>(bias, wih_bf, whh_bf, x_bf, h_bf, out, bar);
}
